// MultiHeadAttention_13649406067534
// MI455X (gfx1250) — matrix operand order repaired and checked by reading, not compiled, not run on hardware
//
#include <hip/hip_runtime.h>

typedef __bf16 bf16;
typedef __attribute__((ext_vector_type(16))) __bf16 v16bf;
typedef __attribute__((ext_vector_type(8)))  __bf16 v8bf;
typedef __attribute__((ext_vector_type(8)))  float  v8f;

#define SHUF16(lo, hi) __builtin_shufflevector(lo, hi, 0,1,2,3,4,5,6,7,8,9,10,11,12,13,14,15)

static constexpr int GEMM_LDST = 40;          // padded LDS stride (halves): 80B rows, 16B-aligned chunks
static constexpr size_t SLAB = 4096u * 1024u; // elements per Q/K/V/attn buffer

// -------------------------------------------------------------------------
// CDNA5 async global->LDS copy (16B per lane, ASYNCcnt-tracked) + wait.
// LDS offset = low 32 bits of the generic pointer (LDS aperture is 4GB-aligned).
// -------------------------------------------------------------------------
__device__ __forceinline__ void async_copy16(const void* gptr, unsigned lds_off) {
    asm volatile("global_load_async_to_lds_b128 %0, %1, off"
                 :: "v"(lds_off), "v"(gptr) : "memory");
}
__device__ __forceinline__ void wait_async0() {
    asm volatile("s_wait_asynccnt 0" ::: "memory");
}
__device__ __forceinline__ unsigned lds_off(const void* p) {
    return (unsigned)(unsigned long long)p;
}

// ---------------------------------------------------------------------------
// Tiled GEMM: C[M,N] = A[M,K] @ B[K,N] + bias[N]
// MODE 0: A fp32, output split into Q(*0.125)/K/V compact bf16 buffers (N=3072)
// MODE 1: A bf16 (async-staged), output fp32 (N=1024)
// Block = 256 threads (8 waves, 4x2), tile 128x128, per-wave 32x64, K-step 32.
// ---------------------------------------------------------------------------
template<int MODE>
__global__ __launch_bounds__(256)
void gemm_wmma_kernel(const void* __restrict__ Av, const float* __restrict__ B,
                      const float* __restrict__ bias, void* __restrict__ Ov,
                      int M, int K, int N)
{
    __shared__ bf16 sA[128 * GEMM_LDST];
    __shared__ bf16 sBT[128 * GEMM_LDST];   // B transposed: sBT[n][k]

    const int tid   = threadIdx.x;
    const int lane  = tid & 31;
    const int wave  = tid >> 5;
    const int lidx  = lane & 15;
    const int lhalf = (lane < 16) ? 0 : 1;
    const int bn = blockIdx.x * 128;
    const int bm = blockIdx.y * 128;
    const int mo = (wave >> 1) * 32;        // 0,32,64,96
    const int no = (wave & 1) * 64;         // 0,64

    v8f acc[2][4];
    for (int i = 0; i < 2; ++i)
        for (int j = 0; j < 4; ++j)
            for (int r = 0; r < 8; ++r) acc[i][j][r] = 0.0f;

    for (int k0 = 0; k0 < K; k0 += 32) {
        __syncthreads();
        // ---- stage A tile 128x32 ----
        if (MODE == 0) {
            const float* A = (const float*)Av;
            for (int i = 0; i < 16; ++i) {
                int idx = tid + i * 256;
                int r = idx >> 5, c = idx & 31;
                sA[r * GEMM_LDST + c] = (bf16)A[(size_t)(bm + r) * K + k0 + c];
            }
        } else {
            const bf16* A = (const bf16*)Av;
            for (int i = 0; i < 2; ++i) {            // 512 x 16B async chunks
                int ch = tid + i * 256;
                int r = ch >> 2, part = ch & 3;
                async_copy16(&A[(size_t)(bm + r) * K + k0 + part * 8],
                             lds_off(&sA[r * GEMM_LDST + part * 8]));
            }
        }
        // ---- stage B tile 32x128 transposed (f32 -> bf16) ----
        for (int i = 0; i < 16; ++i) {
            int idx = tid + i * 256;
            int r = idx >> 7, c = idx & 127;
            sBT[c * GEMM_LDST + r] = (bf16)B[(size_t)(k0 + r) * N + bn + c];
        }
        // ---- prefetch next K tile ----
        if (k0 + 32 < K) {
            if (MODE == 0)
                __builtin_prefetch(&((const float*)Av)[(size_t)(bm + (tid >> 1)) * K + k0 + 32], 0, 0);
            else
                __builtin_prefetch(&((const bf16*)Av)[(size_t)(bm + (tid >> 1)) * K + k0 + 32], 0, 0);
            __builtin_prefetch(&B[(size_t)(k0 + 32 + (tid >> 3)) * N + bn + (tid & 7) * 16], 0, 0);
        }
        if (MODE == 1) wait_async0();
        __syncthreads();

        // A fragments: lane<16 holds K {0..7,16..23}, lane>=16 holds {8..15,24..31}
        v16bf af[2], bfrag[4];
        for (int fm = 0; fm < 2; ++fm) {
            int row = mo + fm * 16 + lidx;
            int kb = lhalf * 8;
            v8bf lo = *(const v8bf*)&sA[row * GEMM_LDST + kb];
            v8bf hi = *(const v8bf*)&sA[row * GEMM_LDST + 16 + kb];
            af[fm] = SHUF16(lo, hi);
        }
        // B fragments: lane<16 col n holds K {0..7,16..23}, lane>=16 holds {8..15,24..31}
        for (int fn = 0; fn < 4; ++fn) {
            int col = no + fn * 16 + lidx;
            int kb = lhalf * 8;
            v8bf lo = *(const v8bf*)&sBT[col * GEMM_LDST + kb];
            v8bf hi = *(const v8bf*)&sBT[col * GEMM_LDST + kb + 16];
            bfrag[fn] = SHUF16(lo, hi);
        }
        for (int fm = 0; fm < 2; ++fm)
            for (int fn = 0; fn < 4; ++fn)
                acc[fm][fn] = __builtin_amdgcn_wmma_f32_16x16x32_bf16(
                    false, af[fm], false, bfrag[fn], (short)0, acc[fm][fn], false, false);
    }

    // epilogue: C layout lane<16 -> M=r, lane>=16 -> M=r+8, N=lane&15
    for (int fm = 0; fm < 2; ++fm)
        for (int fn = 0; fn < 4; ++fn)
            for (int r = 0; r < 8; ++r) {
                int grow = bm + mo + fm * 16 + r + lhalf * 8;
                int gcol = bn + no + fn * 16 + lidx;
                float v = acc[fm][fn][r] + bias[gcol];
                if (MODE == 0) {
                    bf16* ws = (bf16*)Ov;
                    int region = gcol >> 10;        // 0=Q, 1=K, 2=V
                    int cc = gcol & 1023;
                    float scale = (region == 0) ? 0.125f : 1.0f;  // fold 1/sqrt(64) into Q
                    ws[(size_t)region * SLAB + (size_t)grow * 1024 + cc] = (bf16)(v * scale);
                } else {
                    ((float*)Ov)[(size_t)grow * N + gcol] = v;
                }
            }
}

// ---------------------------------------------------------------------------
// Causal flash attention. The reference's RAW reshape makes head (b,h) the
// contiguous [1024,64] slab at (b*16+h)*65536 of the compact Q/K/V buffers,
// and a contiguous output write reproduces the [b,n,e] layout.
// Block = 256 threads (8 waves), 128 queries/block, 64 keys per iteration.
// Q/K tiles staged with CDNA5 async global->LDS copies.
// ---------------------------------------------------------------------------
__global__ __launch_bounds__(256)
void flash_attn_kernel(const bf16* __restrict__ Qb, const bf16* __restrict__ Kb,
                       const bf16* __restrict__ Vb, bf16* __restrict__ Ob)
{
    __shared__ bf16 sQ[128 * 72];
    __shared__ bf16 sK[64 * 72];
    __shared__ bf16 sVT[64 * 72];      // V transposed: sVT[d][k]
    __shared__ bf16 sP[8 * 16 * 72];   // per-wave P scratch (C-layout -> A-layout)

    const int tid   = threadIdx.x;
    const int lane  = tid & 31;
    const int wave  = tid >> 5;
    const int lidx  = lane & 15;
    const int lhalf = (lane < 16) ? 0 : 1;
    const int qbase = blockIdx.x * 128;
    const int bh    = blockIdx.y;           // b*16 + h

    const size_t slab = (size_t)bh * (64u * 1024u);
    const bf16* Qs = Qb + slab;
    const bf16* Ks = Kb + slab;
    const bf16* Vs = Vb + slab;

    // stage Q tile [128][64] via async copies
    for (int i = 0; i < 4; ++i) {
        int ch = tid + i * 256;             // 1024 chunks of 8 halves
        int q = ch >> 3, part = ch & 7;
        async_copy16(&Qs[(qbase + q) * 64 + part * 8],
                     lds_off(&sQ[q * 72 + part * 8]));
    }
    wait_async0();
    __syncthreads();

    // per-wave Q A-fragments, one per 32-wide K(d) chunk
    v16bf aq[2];
    for (int c = 0; c < 2; ++c) {
        int row = wave * 16 + lidx;
        int d0 = c * 32 + lhalf * 8;
        v8bf lo = *(const v8bf*)&sQ[row * 72 + d0];
        v8bf hi = *(const v8bf*)&sQ[row * 72 + d0 + 16];
        aq[c] = SHUF16(lo, hi);
    }

    v8f o[4];
    float m[8], l[8];
    for (int j = 0; j < 4; ++j)
        for (int r = 0; r < 8; ++r) o[j][r] = 0.0f;
    for (int r = 0; r < 8; ++r) { m[r] = -1e30f; l[r] = 0.0f; }

    const int nkb = (qbase + 128) >> 6;     // key blocks needed for causal coverage
    for (int kbi = 0; kbi < nkb; ++kbi) {
        const int k0 = kbi * 64;
        __syncthreads();
        // stage K [64][64] row-major via async copies
        for (int i = 0; i < 2; ++i) {
            int ch = tid + i * 256;
            int k = ch >> 3, part = ch & 7;
            async_copy16(&Ks[(k0 + k) * 64 + part * 8],
                         lds_off(&sK[k * 72 + part * 8]));
        }
        // stage V transposed (u32 loads -> two scalar LDS stores)
        for (int i = 0; i < 8; ++i) {
            int idx = tid + i * 256;        // 2048 u32 pairs
            int k = idx >> 5, d2 = (idx & 31) * 2;
            union { unsigned u; bf16 h[2]; } cv;
            cv.u = *(const unsigned*)&Vs[(k0 + k) * 64 + d2];
            sVT[d2 * 72 + k]       = cv.h[0];
            sVT[(d2 + 1) * 72 + k] = cv.h[1];
        }
        wait_async0();
        __syncthreads();

        // S = Q @ K^T (softmax scale already folded into Q)
        v8f s[4];
        for (int j = 0; j < 4; ++j)
            for (int r = 0; r < 8; ++r) s[j][r] = 0.0f;
        for (int c = 0; c < 2; ++c) {
            int d0 = c * 32 + lhalf * 8;
            for (int j = 0; j < 4; ++j) {
                int kcol = j * 16 + lidx;
                v8bf lo = *(const v8bf*)&sK[kcol * 72 + d0];
                v8bf hi = *(const v8bf*)&sK[kcol * 72 + d0 + 16];
                v16bf bk = SHUF16(lo, hi);
                s[j] = __builtin_amdgcn_wmma_f32_16x16x32_bf16(
                    false, aq[c], false, bk, (short)0, s[j], false, false);
            }
        }

        // causal mask (only near the diagonal)
        if (k0 + 63 > qbase + wave * 16) {
            for (int j = 0; j < 4; ++j)
                for (int r = 0; r < 8; ++r) {
                    int q = qbase + wave * 16 + r + lhalf * 8;
                    int k = k0 + j * 16 + lidx;
                    if (k > q) s[j][r] = -1e30f;
                }
        }

        // online softmax (rows live across lanes 0..15 / 16..31 per C layout)
        float mn[8], alpha[8], rs[8];
        for (int r = 0; r < 8; ++r) {
            float t = fmaxf(fmaxf(s[0][r], s[1][r]), fmaxf(s[2][r], s[3][r]));
            for (int off = 8; off >= 1; off >>= 1)
                t = fmaxf(t, __shfl_xor(t, off, 16));
            mn[r] = fmaxf(m[r], t);
            alpha[r] = __expf(m[r] - mn[r]);
            m[r] = mn[r];
            rs[r] = 0.0f;
        }
        for (int j = 0; j < 4; ++j)
            for (int r = 0; r < 8; ++r) {
                float p = __expf(s[j][r] - mn[r]);
                s[j][r] = p;
                rs[r] += p;
            }
        for (int r = 0; r < 8; ++r) {
            float t = rs[r];
            for (int off = 8; off >= 1; off >>= 1)
                t += __shfl_xor(t, off, 16);
            l[r] = l[r] * alpha[r] + t;
        }
        for (int j = 0; j < 4; ++j)
            for (int r = 0; r < 8; ++r) o[j][r] *= alpha[r];

        // P -> per-wave LDS (C layout -> row-major) for A-fragment loads
        bf16* sPw = &sP[wave * 16 * 72];
        for (int j = 0; j < 4; ++j)
            for (int r = 0; r < 8; ++r) {
                int row = r + lhalf * 8;
                sPw[row * 72 + j * 16 + lidx] = (bf16)s[j][r];
            }
        __syncthreads();

        // O += P @ V
        for (int c = 0; c < 2; ++c) {
            int d0 = c * 32 + lhalf * 8;
            v8bf plo = *(const v8bf*)&sPw[lidx * 72 + d0];
            v8bf phi = *(const v8bf*)&sPw[lidx * 72 + d0 + 16];
            v16bf ap = SHUF16(plo, phi);
            int kk = c * 32 + lhalf * 8;
            for (int j = 0; j < 4; ++j) {
                int dcol = j * 16 + lidx;
                v8bf blo = *(const v8bf*)&sVT[dcol * 72 + kk];
                v8bf bhi = *(const v8bf*)&sVT[dcol * 72 + kk + 16];
                v16bf bv = SHUF16(blo, bhi);
                o[j] = __builtin_amdgcn_wmma_f32_16x16x32_bf16(
                    false, ap, false, bv, (short)0, o[j], false, false);
            }
        }
    }

    // normalize + contiguous store (reproduces [b,n,e] layout)
    for (int r = 0; r < 8; ++r) l[r] = 1.0f / l[r];
    for (int j = 0; j < 4; ++j)
        for (int r = 0; r < 8; ++r) {
            int q = qbase + wave * 16 + r + lhalf * 8;
            int d = j * 16 + lidx;
            Ob[slab + (size_t)q * 64 + d] = (bf16)(o[j][r] * l[r]);
        }
}

// ---------------------------------------------------------------------------
extern "C" void kernel_launch(void* const* d_in, const int* in_sizes, int n_in,
                              void* d_out, int out_size, void* d_ws, size_t ws_size,
                              hipStream_t stream)
{
    const float* x    = (const float*)d_in[0];   // [4096, 1024]
    const float* Wqkv = (const float*)d_in[1];   // [1024, 3072]
    const float* bqkv = (const float*)d_in[2];   // [3072]
    const float* Wfc  = (const float*)d_in[3];   // [1024, 1024]
    const float* bfc  = (const float*)d_in[4];   // [1024]
    float* out = (float*)d_out;                  // [4096, 1024]

    bf16* Qb   = (bf16*)d_ws;                    // compact bf16 buffers
    bf16* Kb   = Qb + SLAB;
    bf16* Vb   = Qb + 2 * SLAB;
    bf16* attn = Qb + 3 * SLAB;                  // 32 MB total workspace

    dim3 gQKV(3072 / 128, 4096 / 128);
    gemm_wmma_kernel<0><<<gQKV, 256, 0, stream>>>(x, Wqkv, bqkv, (void*)Qb, 4096, 1024, 3072);

    dim3 gAtt(1024 / 128, 4 * 16);
    flash_attn_kernel<<<gAtt, 256, 0, stream>>>(Qb, Kb, Vb, attn);

    dim3 gFC(1024 / 128, 4096 / 128);
    gemm_wmma_kernel<1><<<gFC, 256, 0, stream>>>(attn, Wfc, bfc, (void*)out, 4096, 1024, 1024);
}
